// GatedDeltaNet_70738111365308
// MI455X (gfx1250) — matrix operand order repaired and checked by reading, not compiled, not run on hardware
//
// GatedDeltaNet forward for MI455X (gfx1250, wave32, WMMA + async LDS copies).
// Pipeline: [bf16 WMMA GEMM projections] -> [gk/beta] -> [conv+SiLU+L2 head split]
//        -> [delta_prep: KK/attn via bf16 WMMA + triangular solves in LDS]
//        -> [delta_scan: fp32-state recurrence, f32 WMMA, S resident in 186KB LDS]
//        -> [gated RMSNorm] -> [bf16 WMMA output GEMM].
#include <hip/hip_runtime.h>

#define DM   1024
#define KD   768
#define VD   1536
#define NH   8
#define HQK  96
#define HV   192
#define CH   64
#define BBS  2
#define LLS  4096
#define NCH  (LLS/CH)    // 64 chunks
#define BL   (BBS*LLS)   // 8192 tokens
#define BHN  (BBS*NH)    // 16 (b,h) chains

typedef __attribute__((ext_vector_type(16))) __bf16 v16bf;
typedef __attribute__((ext_vector_type(8)))  float  v8f;
typedef __attribute__((ext_vector_type(2)))  float  v2f;

__device__ __forceinline__ __bf16 f2bf(float f) {
  unsigned u = __builtin_bit_cast(unsigned, f);
  u += 0x7FFFu + ((u >> 16) & 1u);                 // round-to-nearest-even
  unsigned short h = (unsigned short)(u >> 16);
  return __builtin_bit_cast(__bf16, h);
}
__device__ __forceinline__ float bf2f(__bf16 b) {
  unsigned short h = __builtin_bit_cast(unsigned short, b);
  return __builtin_bit_cast(float, ((unsigned)h) << 16);
}
__device__ __forceinline__ float sigm(float x) { return 1.f / (1.f + __expf(-x)); }
__device__ __forceinline__ float silu(float x) { return x * sigm(x); }

// ---- async global->LDS copy (gfx1250 ASYNCcnt path, CDNA5 ISA) ----
// lds_off: byte address within the wave's LDS allocation (dynamic LDS starts at 0 here).
__device__ __forceinline__ void async_g2l_b128(unsigned lds_off, const void* g) {
  unsigned long long ga = (unsigned long long)g;
  asm volatile("global_load_async_to_lds_b128 %0, %1, off"
               :: "v"(lds_off), "v"(ga) : "memory");
}
__device__ __forceinline__ void wait_async0() {
  asm volatile("s_wait_asynccnt 0" ::: "memory");
}

// ---------- WMMA fragment helpers (gfx1250 wave32 layouts, CDNA5 ISA) ----------
// A (16x32 bf16): lane<16 holds K 0-7 & 16-23 of row M=lane; lane>=16 holds K 8-15 & 24-31.
__device__ __forceinline__ v16bf frag_a_bf16(const __bf16* t, int ld, int m0, int k0, int lane) {
  int row = m0 + (lane & 15), hs = (lane >> 4) * 8;
  const __bf16* p = t + (size_t)row * ld + k0;
  v16bf a;
#pragma unroll
  for (int j = 0; j < 8; ++j) a[j] = p[hs + j];
#pragma unroll
  for (int j = 0; j < 8; ++j) a[8 + j] = p[16 + hs + j];
  return a;
}
// B = tile^T where tile is row-major [N][ld]: lane<16 holds K 0-7 & 16-23, lane>=16 K 8-15 & 24-31 -> 2x ds_load_b128
__device__ __forceinline__ v16bf frag_bT_bf16(const __bf16* t, int ld, int n0, int k0, int lane) {
  int col = n0 + (lane & 15), hs = (lane >> 4) * 8;
  const __bf16* p = t + (size_t)col * ld + k0 + hs;
  v16bf b;
#pragma unroll
  for (int h = 0; h < 8; ++h) b[h] = p[h];
#pragma unroll
  for (int h = 0; h < 8; ++h) b[8 + h] = p[16 + h];
  return b;
}
// C/D (16x16 f32): VGPR r -> M = m0 + r + 8*(lane>=16), N = n0 + (lane&15)
__device__ __forceinline__ void store_c(float* t, int ld, int m0, int n0, int lane, v8f c) {
  int n = n0 + (lane & 15), mo = (lane >> 4) * 8;
#pragma unroll
  for (int r = 0; r < 8; ++r) t[(size_t)(m0 + mo + r) * ld + n] = c[r];
}
__device__ __forceinline__ v8f vzero8() {
  v8f c;
#pragma unroll
  for (int r = 0; r < 8; ++r) c[r] = 0.f;
  return c;
}

// ================= generic GEMM: C[M,N] = A[M,K](f32) * W[K,N](f32), bf16 WMMA =================
__global__ __launch_bounds__(128) void gemm_bf16_wmma(
    const float* __restrict__ A, const float* __restrict__ W,
    float* __restrict__ C, int M, int N, int K) {
  __shared__ __bf16 sA[64 * 32];    // [m][k]
  __shared__ __bf16 sBT[64 * 32];   // [n][k]  (transposed: B fragments are contiguous)
  int tid = threadIdx.x, lane = tid & 31, wid = tid >> 5;
  int m0 = blockIdx.y * 64, n0 = blockIdx.x * 64;
  int wm = (wid >> 1) * 32, wn = (wid & 1) * 32;   // 4 waves -> 2x2 of 32x32
  v8f acc[2][2];
  for (int i = 0; i < 2; ++i) for (int j = 0; j < 2; ++j) acc[i][j] = vzero8();
  for (int k0 = 0; k0 < K; k0 += 32) {
    for (int i = tid; i < 64 * 32; i += 128) {
      int r = i >> 5, c = i & 31;
      sA[i] = f2bf(A[(size_t)(m0 + r) * K + (k0 + c)]);
    }
    for (int i = tid; i < 32 * 64; i += 128) {
      int r = i >> 6, c = i & 63;                                  // r=k, c=n (coalesced read)
      sBT[c * 32 + r] = f2bf(W[(size_t)(k0 + r) * N + (n0 + c)]);
    }
    if (k0 + 32 < K) {  // prefetch next K-tile -> global_prefetch_b8
      __builtin_prefetch(&A[(size_t)(m0 + (tid >> 1)) * K + k0 + 32], 0, 1);
      __builtin_prefetch(&W[(size_t)(k0 + 32 + (tid >> 3)) * N + n0], 0, 1);
    }
    __syncthreads();
    v16bf bfr[2];
#pragma unroll
    for (int ni = 0; ni < 2; ++ni) bfr[ni] = frag_bT_bf16(sBT, 32, wn + ni * 16, 0, lane);
#pragma unroll
    for (int mi = 0; mi < 2; ++mi) {
      v16bf a = frag_a_bf16(sA, 32, wm + mi * 16, 0, lane);
#pragma unroll
      for (int ni = 0; ni < 2; ++ni)
        acc[mi][ni] = __builtin_amdgcn_wmma_f32_16x16x32_bf16(
            false, a, false, bfr[ni], (short)0, acc[mi][ni], false, false);
    }
    __syncthreads();
  }
#pragma unroll
  for (int mi = 0; mi < 2; ++mi)
#pragma unroll
    for (int ni = 0; ni < 2; ++ni)
      store_c(C + (size_t)m0 * N + n0, N, wm + mi * 16, wn + ni * 16, lane, acc[mi][ni]);
}

// ================= gk / beta small projections + activations =================
__global__ __launch_bounds__(256) void gkbeta_kernel(
    const float* __restrict__ u, const float* __restrict__ Wgk, const float* __restrict__ Wb,
    const float* __restrict__ b_b, const float* __restrict__ A_log, const float* __restrict__ dt_bias,
    float* __restrict__ gk_o, float* __restrict__ beta_o) {
  int idx = blockIdx.x * 256 + threadIdx.x;
  if (idx >= BL * NH) return;
  int tok = idx >> 3, h = idx & 7;
  const float* ur = u + (size_t)tok * DM;
  float s1 = 0.f, s2 = 0.f;
  for (int i = 0; i < DM; ++i) {
    float uv = ur[i];
    s1 = fmaf(uv, Wgk[i * NH + h], s1);
    s2 = fmaf(uv, Wb[i * NH + h], s2);
  }
  float x = s1 + dt_bias[h];
  float sp = (x > 20.f) ? x : log1pf(__expf(x));
  int b = tok / LLS, l = tok % LLS;
  size_t o = (size_t)(b * NH + h) * LLS + l;
  gk_o[o]   = -__expf(A_log[h]) * sp;
  beta_o[o] = sigm(s2 + b_b[h]);
}

// ============ causal dwconv(4) + SiLU (+ per-head L2 norm) + (B,H,L,Dh) bf16 split ============
__global__ __launch_bounds__(256) void conv_silu_kernel(
    const float* __restrict__ x, const float* __restrict__ wc,
    __bf16* __restrict__ out, int C, int Dh, int do_l2, float scale) {
  __shared__ float sbuf[VD];
  __shared__ float sscale[NH];
  int l = blockIdx.x, b = blockIdx.y, tid = threadIdx.x;
  for (int c = tid; c < C; c += 256) {
    float acc = 0.f;
#pragma unroll
    for (int j = 0; j < 4; ++j) {
      int ls = l - 3 + j;
      if (ls >= 0) acc = fmaf(x[((size_t)b * LLS + ls) * C + c], wc[c * 4 + j], acc);
    }
    sbuf[c] = silu(acc);
  }
  __syncthreads();
  if (tid < NH) {
    if (do_l2) {
      float ss = 0.f;
      for (int d = 0; d < Dh; ++d) { float v = sbuf[tid * Dh + d]; ss = fmaf(v, v, ss); }
      sscale[tid] = scale / fmaxf(sqrtf(ss), 1e-12f);
    } else sscale[tid] = scale;
  }
  __syncthreads();
  for (int c = tid; c < C; c += 256) {
    int h = c / Dh, d = c - h * Dh;
    out[((size_t)(b * NH + h) * LLS + l) * Dh + d] = f2bf(sbuf[c] * sscale[h]);
  }
}

// ============== per-chunk prep: KK & attn via bf16 WMMA, triangular solves in LDS ==============
__global__ __launch_bounds__(256) void delta_prep(
    const __bf16* __restrict__ qg, const __bf16* __restrict__ kg, const __bf16* __restrict__ vg,
    const float* __restrict__ gkg, const float* __restrict__ betag,
    float* __restrict__ ug, float* __restrict__ kcdg, float* __restrict__ attng,
    float* __restrict__ decg) {
  extern __shared__ char smem[];
  float*  KK     = (float*)smem;           // 64*64
  float*  RU     = KK + 64 * 64;           // 64*192  (v*beta -> u_)
  float*  RK     = RU + 64 * 192;          // 64*96   (k*beta -> kcd)
  float*  dec_s  = RK + 64 * 96;           // 64
  float*  beta_s = dec_s + 64;             // 64
  __bf16* qt     = (__bf16*)(beta_s + 64); // 64*96
  __bf16* kt     = qt + 64 * 96;           // 64*96
  __bf16* kbt    = kt + 64 * 96;           // 64*96
  int n = blockIdx.x, bh = blockIdx.y;
  int tid = threadIdx.x, lane = tid & 31, wid = tid >> 5;
  size_t rb = (size_t)bh * LLS + (size_t)n * CH;

  // async direct copies of q/k chunk tiles (3x b128 per thread, uniform EXEC)
  {
    unsigned oq = (unsigned)((char*)qt - smem), ok = (unsigned)((char*)kt - smem);
    const char* gq = (const char*)(qg + rb * HQK);
    const char* gk = (const char*)(kg + rb * HQK);
    for (int i = tid * 16; i < CH * HQK * 2; i += 256 * 16) {
      async_g2l_b128(oq + i, gq + i);
      async_g2l_b128(ok + i, gk + i);
    }
  }
  if (tid < CH) {
    beta_s[tid] = betag[rb + tid];
    float s = 0.f;
    for (int j = 0; j <= tid; ++j) s += gkg[rb + j];   // chunk-local cumsum
    dec_s[tid] = s;
    decg[rb + tid] = s;
  }
  for (int i = tid; i < CH * HV; i += 256) RU[i] = bf2f(vg[rb * HV + i]);  // *beta below
  wait_async0();
  __syncthreads();
  for (int i = tid; i < CH * HQK; i += 256) {
    int r = i / HQK;
    kbt[i] = f2bf(bf2f(kt[i]) * beta_s[r]);
    RK[i]  = bf2f(kt[i]) * beta_s[r];
  }
  for (int i = tid; i < CH * HV; i += 256) { int r = i / HV; RU[i] *= beta_s[r]; }
  __syncthreads();

  // KK = (beta*k) @ k^T  (64x64, K=96)
  for (int t = wid; t < 16; t += 8) {
    int m0 = (t >> 2) * 16, n0 = (t & 3) * 16;
    v8f acc = vzero8();
    for (int k0 = 0; k0 < HQK; k0 += 32) {
      v16bf a = frag_a_bf16(kbt, HQK, m0, k0, lane);
      v16bf b = frag_bT_bf16(kt, HQK, n0, k0, lane);
      acc = __builtin_amdgcn_wmma_f32_16x16x32_bf16(false, a, false, b, (short)0, acc, false, false);
    }
    store_c(KK, CH, m0, n0, lane, acc);
  }
  __syncthreads();

  // combined unit-lower-triangular forward substitution for u_ (M1) and kcd (M2)
  for (int i = 0; i < CH - 1; ++i) {
    int rows = CH - 1 - i;
    for (int t = tid; t < rows * (HV + HQK); t += 256) {
      int j = i + 1 + t / (HV + HQK);
      int c = t % (HV + HQK);
      float kk = KK[j * CH + i];
      if (c < HV) {
        RU[j * HV + c] -= kk * __expf(fminf(dec_s[j] - dec_s[i], 0.f)) * RU[i * HV + c];
      } else {
        int d = c - HV;
        RK[j * HQK + d] -= kk * RK[i * HQK + d];
      }
    }
    __syncthreads();
  }
  for (int i = tid; i < CH * HV;  i += 256) ug[rb * HV + i]    = RU[i];
  for (int i = tid; i < CH * HQK; i += 256) kcdg[rb * HQK + i] = RK[i];

  // attn = tril(q @ k^T) * Lmask (64x64, K=96)
  for (int t = wid; t < 16; t += 8) {
    int m0 = (t >> 2) * 16, n0 = (t & 3) * 16;
    v8f acc = vzero8();
    for (int k0 = 0; k0 < HQK; k0 += 32) {
      v16bf a = frag_a_bf16(qt, HQK, m0, k0, lane);
      v16bf b = frag_bT_bf16(kt, HQK, n0, k0, lane);
      acc = __builtin_amdgcn_wmma_f32_16x16x32_bf16(false, a, false, b, (short)0, acc, false, false);
    }
    int jn = n0 + (lane & 15), mo = (lane >> 4) * 8;
#pragma unroll
    for (int r = 0; r < 8; ++r) {
      int im = m0 + mo + r;
      float val = (jn <= im) ? acc[r] * __expf(fminf(dec_s[im] - dec_s[jn], 0.f)) : 0.f;
      attng[(((size_t)bh * NCH + n) * CH + im) * CH + jn] = val;
    }
  }
}

// ========= sequential state scan: S(96x192,f32) resident in LDS; f32 WMMA (16x16x4) =========
// LDS layouts: ST[e][d] 192x97 (padded), VNT[e][i] 192x65, ktT[d][i] 96x65 -> all f32-WMMA
// A/B fetches are contiguous pairs; S read-modify-write is 8 contiguous floats per lane.
__global__ __launch_bounds__(256) void delta_scan(
    const __bf16* __restrict__ qg, const __bf16* __restrict__ kg,
    const float* __restrict__ ug, const float* __restrict__ kcdg,
    const float* __restrict__ attng, const float* __restrict__ decg,
    float* __restrict__ og) {
  extern __shared__ char smem[];
  float*  ST    = (float*)smem;          // [192][97] state transposed (e,d)
  float*  VNT   = ST + 192 * 97;         // [192][65] u_ -> v_new transposed (e,i)
  float*  KC    = VNT + 192 * 65;        // [64][96]
  float*  AT    = KC + 64 * 96;          // [64][64]
  float*  dec_s = AT + 64 * 64;          // 64
  __bf16* ktT   = (__bf16*)(dec_s + 64); // [96][65] k transposed (d,i)
  __bf16* qt    = ktT + 96 * 65;         // [64][96]
  int bh = blockIdx.x, tid = threadIdx.x, lane = tid & 31, wid = tid >> 5;
  for (int i = tid; i < 192 * 97; i += 256) ST[i] = 0.f;

  for (int n = 0; n < NCH; ++n) {
    __syncthreads();
    size_t rb = (size_t)bh * LLS + (size_t)n * CH;
    // async direct copies (uniform EXEC: 6 + 4 + 3 b128 per thread)
    {
      unsigned oKC = (unsigned)((char*)KC - smem);
      const char* gKC = (const char*)(kcdg + rb * HQK);
      for (int i = tid * 16; i < CH * HQK * 4; i += 256 * 16) async_g2l_b128(oKC + i, gKC + i);
      unsigned oAT = (unsigned)((char*)AT - smem);
      const char* gAT = (const char*)(attng + ((size_t)bh * NCH + n) * (CH * CH));
      for (int i = tid * 16; i < CH * CH * 4; i += 256 * 16) async_g2l_b128(oAT + i, gAT + i);
      unsigned oQ = (unsigned)((char*)qt - smem);
      const char* gQ = (const char*)(qg + rb * HQK);
      for (int i = tid * 16; i < CH * HQK * 2; i += 256 * 16) async_g2l_b128(oQ + i, gQ + i);
    }
    // transposed copies (coalesced global reads; padded LDS -> conflict-free)
    for (int i = tid; i < CH * HV; i += 256) {
      int r = i / HV, e = i - r * HV;
      VNT[e * 65 + r] = ug[rb * HV + i];
    }
    for (int i = tid; i < CH * HQK; i += 256) {
      int r = i / HQK, d = i - r * HQK;
      ktT[d * 65 + r] = kg[rb * HQK + i];
    }
    if (tid < CH) dec_s[tid] = decg[rb + tid];
    wait_async0();
    __syncthreads();

    // v_new(i,e) = u_(i,e) - (kcd * e^dec)(i,d) @ S(d,e)     (64x192, K=96)
    for (int t = wid; t < 48; t += 8) {
      int m0 = (t / 12) * 16, e0 = (t % 12) * 16;
      int row = m0 + (lane & 15), hs2 = (lane >> 4) * 2, mo = (lane >> 4) * 8;
      int ncol = e0 + (lane & 15);
      float neg_ed = -__expf(dec_s[row]);
      v8f acc;
#pragma unroll
      for (int r = 0; r < 8; ++r) acc[r] = VNT[(size_t)ncol * 65 + m0 + mo + r];
      for (int d = 0; d < HQK; d += 4) {
        v2f a, b;
        a[0] = KC[row * HQK + d + hs2]     * neg_ed;
        a[1] = KC[row * HQK + d + hs2 + 1] * neg_ed;
        b[0] = ST[(size_t)ncol * 97 + d + hs2];
        b[1] = ST[(size_t)ncol * 97 + d + hs2 + 1];
        acc = __builtin_amdgcn_wmma_f32_16x16x4_f32(false, a, false, b, (short)0, acc, false, false);
      }
#pragma unroll
      for (int r = 0; r < 8; ++r) VNT[(size_t)ncol * 65 + m0 + mo + r] = acc[r];
    }
    __syncthreads();

    // o(i,e) = (q * e^dec)(i,d) @ S(d,e) + attn(i,j) @ v_new(j,e)
    for (int t = wid; t < 48; t += 8) {
      int m0 = (t / 12) * 16, e0 = (t % 12) * 16;
      int row = m0 + (lane & 15), hs2 = (lane >> 4) * 2, mo = (lane >> 4) * 8;
      int ncol = e0 + (lane & 15);
      float edr = __expf(dec_s[row]);
      v8f acc = vzero8();
      for (int d = 0; d < HQK; d += 4) {
        v2f a, b;
        a[0] = bf2f(qt[row * HQK + d + hs2])     * edr;
        a[1] = bf2f(qt[row * HQK + d + hs2 + 1]) * edr;
        b[0] = ST[(size_t)ncol * 97 + d + hs2];
        b[1] = ST[(size_t)ncol * 97 + d + hs2 + 1];
        acc = __builtin_amdgcn_wmma_f32_16x16x4_f32(false, a, false, b, (short)0, acc, false, false);
      }
      for (int i2 = 0; i2 < CH; i2 += 4) {
        v2f a, b;
        a[0] = AT[row * CH + i2 + hs2];
        a[1] = AT[row * CH + i2 + hs2 + 1];
        b[0] = VNT[(size_t)ncol * 65 + i2 + hs2];
        b[1] = VNT[(size_t)ncol * 65 + i2 + hs2 + 1];
        acc = __builtin_amdgcn_wmma_f32_16x16x4_f32(false, a, false, b, (short)0, acc, false, false);
      }
#pragma unroll
      for (int r = 0; r < 8; ++r) og[(rb + m0 + mo + r) * HV + ncol] = acc[r];
    }
    __syncthreads();

    // S(d,e) = S(d,e)*e^dl + (k * e^(dl-dec))^T(d,i) @ v_new(i,e)   (96x192, K=64)
    float dl = dec_s[63];
    float edl = __expf(dl);
    for (int t = wid; t < 72; t += 8) {
      int d0 = (t / 12) * 16, e0 = (t % 12) * 16;
      int drow = d0 + (lane & 15), hs2 = (lane >> 4) * 2, mo = (lane >> 4) * 8;
      int ncol = e0 + (lane & 15);
      v8f acc = vzero8();
      for (int i2 = 0; i2 < CH; i2 += 4) {
        int ka = i2 + hs2;
        v2f a, b;
        a[0] = bf2f(ktT[(size_t)drow * 65 + ka])     * __expf(dl - dec_s[ka]);
        a[1] = bf2f(ktT[(size_t)drow * 65 + ka + 1]) * __expf(dl - dec_s[ka + 1]);
        b[0] = VNT[(size_t)ncol * 65 + ka];
        b[1] = VNT[(size_t)ncol * 65 + ka + 1];
        acc = __builtin_amdgcn_wmma_f32_16x16x4_f32(false, a, false, b, (short)0, acc, false, false);
      }
#pragma unroll
      for (int r = 0; r < 8; ++r) {
        size_t si = (size_t)ncol * 97 + d0 + mo + r;
        ST[si] = ST[si] * edl + acc[r];
      }
    }
  }
}

// ================= gated RMSNorm: gated = rmsnorm(o)*w*silu(g), (B,L,1536) =================
__global__ __launch_bounds__(256) void gate_kernel(
    const float* __restrict__ o, const float* __restrict__ g,
    const float* __restrict__ nw, float* __restrict__ gated) {
  __shared__ float so[VD];
  __shared__ float sr[NH];
  int l = blockIdx.x, b = blockIdx.y, tid = threadIdx.x;
  for (int c = tid; c < VD; c += 256) {
    int h = c / HV, d = c - h * HV;
    so[c] = o[((size_t)(b * NH + h) * LLS + l) * HV + d];
  }
  __syncthreads();
  if (tid < NH) {
    float ss = 0.f;
    for (int d = 0; d < HV; ++d) { float v = so[tid * HV + d]; ss = fmaf(v, v, ss); }
    sr[tid] = rsqrtf(ss / (float)HV + 1e-5f);
  }
  __syncthreads();
  size_t rowo = ((size_t)b * LLS + l) * VD;
  for (int c = tid; c < VD; c += 256) {
    int h = c / HV, d = c - h * HV;
    float gg = g[rowo + c];
    gated[rowo + c] = so[c] * sr[h] * nw[d] * silu(gg);
  }
}

extern "C" void kernel_launch(void* const* d_in, const int* in_sizes, int n_in,
                              void* d_out, int out_size, void* d_ws, size_t ws_size,
                              hipStream_t stream) {
  (void)in_sizes; (void)n_in; (void)out_size; (void)ws_size;
  const float* u     = (const float*)d_in[0];
  const float* Wq    = (const float*)d_in[1];
  const float* Wk    = (const float*)d_in[2];
  const float* Wv    = (const float*)d_in[3];
  const float* Wg    = (const float*)d_in[4];
  const float* Wo    = (const float*)d_in[5];
  const float* Wgk   = (const float*)d_in[6];
  const float* Wb    = (const float*)d_in[7];
  const float* b_b   = (const float*)d_in[8];
  const float* A_log = (const float*)d_in[9];
  const float* dt_b  = (const float*)d_in[10];
  const float* cq    = (const float*)d_in[11];
  const float* ck    = (const float*)d_in[12];
  const float* cv    = (const float*)d_in[13];
  const float* nw    = (const float*)d_in[14];
  float* out = (float*)d_out;

  char* ws = (char*)d_ws;
  size_t off = 0;
  auto take = [&](size_t bytes) -> char* {
    char* p = ws + off;
    off += (bytes + 255) & ~(size_t)255;
    return p;
  };
  float*  q_pre = (float*)take((size_t)BL * KD * 4);   // reused as gated[0:25MB]
  float*  k_pre = (float*)take((size_t)BL * KD * 4);   // reused as gated[25MB:50MB]
  float*  v_pre = (float*)take((size_t)BL * VD * 4);   // reused as o buffer
  float*  g_buf = (float*)take((size_t)BL * VD * 4);
  __bf16* q_bf  = (__bf16*)take((size_t)BL * KD * 2);  // (B,H,L,96)
  __bf16* k_bf  = (__bf16*)take((size_t)BL * KD * 2);
  __bf16* v_bf  = (__bf16*)take((size_t)BL * VD * 2);  // (B,H,L,192)
  float*  gk_b  = (float*)take((size_t)BL * NH * 4);   // (B,H,L)
  float*  be_b  = (float*)take((size_t)BL * NH * 4);
  float*  dec_b = (float*)take((size_t)BL * NH * 4);
  float*  u_b   = (float*)take((size_t)BL * VD * 4);   // u_ (B,H,L,192)
  float*  kcd_b = (float*)take((size_t)BL * KD * 4);   // kcd (B,H,L,96)
  float*  at_b  = (float*)take((size_t)BHN * NCH * CH * CH * 4);
  float*  gated = q_pre;   // 2*BL*KD*4 == BL*VD*4 bytes, contiguous
  float*  o_buf = v_pre;   // same size, lifetime disjoint

  const int PREP_LDS = (64 * 64 + 64 * 192 + 64 * 96 + 128) * 4 + 3 * (64 * 96) * 2;          // 127488
  const int SCAN_LDS = (192 * 97 + 192 * 65 + 64 * 96 + 64 * 64 + 64) * 4 + (96 * 65 + 64 * 96) * 2; // 190400
  (void)hipFuncSetAttribute((const void*)delta_prep, hipFuncAttributeMaxDynamicSharedMemorySize, PREP_LDS);
  (void)hipFuncSetAttribute((const void*)delta_scan, hipFuncAttributeMaxDynamicSharedMemorySize, SCAN_LDS);

  dim3 blk128(128), blk256(256);
  // projections (bf16 WMMA)
  gemm_bf16_wmma<<<dim3(KD / 64, BL / 64), blk128, 0, stream>>>(u, Wq, q_pre, BL, KD, DM);
  gemm_bf16_wmma<<<dim3(KD / 64, BL / 64), blk128, 0, stream>>>(u, Wk, k_pre, BL, KD, DM);
  gemm_bf16_wmma<<<dim3(VD / 64, BL / 64), blk128, 0, stream>>>(u, Wv, v_pre, BL, VD, DM);
  gemm_bf16_wmma<<<dim3(VD / 64, BL / 64), blk128, 0, stream>>>(u, Wg, g_buf, BL, VD, DM);
  gkbeta_kernel<<<dim3((BL * NH + 255) / 256), blk256, 0, stream>>>(u, Wgk, Wb, b_b, A_log, dt_b, gk_b, be_b);
  // conv + SiLU (+L2 per head); q also folds dk^-0.5 = 96^-0.5
  conv_silu_kernel<<<dim3(LLS, BBS), blk256, 0, stream>>>(q_pre, cq, q_bf, KD, HQK, 1, 0.102062072616f);
  conv_silu_kernel<<<dim3(LLS, BBS), blk256, 0, stream>>>(k_pre, ck, k_bf, KD, HQK, 1, 1.f);
  conv_silu_kernel<<<dim3(LLS, BBS), blk256, 0, stream>>>(v_pre, cv, v_bf, VD, HV, 0, 1.f);
  // chunked gated delta rule
  delta_prep<<<dim3(NCH, BHN), blk256, PREP_LDS, stream>>>(q_bf, k_bf, v_bf, gk_b, be_b, u_b, kcd_b, at_b, dec_b);
  delta_scan<<<dim3(BHN), blk256, SCAN_LDS, stream>>>(q_bf, k_bf, u_b, kcd_b, at_b, dec_b, o_buf);
  // gated RMSNorm + output projection
  gate_kernel<<<dim3(LLS, BBS), blk256, 0, stream>>>(o_buf, g_buf, nw, gated);
  gemm_bf16_wmma<<<dim3(DM / 64, BL / 64), blk128, 0, stream>>>(gated, Wo, out, BL, DM, VD);
}
